// CogKR_65352222376847
// MI455X (gfx1250) — compile-verified, matrix operand order checked by reading, not run on hardware
//
#include <hip/hip_runtime.h>

// ---------------------------------------------------------------------------
// CDNA5 (gfx1250, wave32) implementation of CogKR GRU aggregation.
//
// Factorization: gi/gh gate pre-activations depend only on table rows, so we
// precompute three dense bf16-WMMA GEMMs (G_rel, G_ent, G_h) and turn the
// per-(b,t,m) work into pure gathers + GRU elementwise + masked mean.
// ---------------------------------------------------------------------------

typedef __attribute__((ext_vector_type(16))) __bf16 v16bf;
typedef __attribute__((ext_vector_type(8)))  __bf16 v8bf;
typedef __attribute__((ext_vector_type(8)))  float  v8f;

#define LDS_STRIDE 136   // 128 bf16 cols + 8 bf16 pad -> 272B row stride (bank-conflict free)

// Load a 16-lane-row x 32-K bf16 fragment from an LDS row (row-major, LDS_STRIDE)
// following the CDNA5 16-bit A/B VGPR layout:
//   lanes 0-15  hold K = k0+0..7  then k0+16..23
//   lanes 16-31 hold K = k0+8..15 then k0+24..31
__device__ __forceinline__ v16bf lds_frag(const __bf16* rowBase, int k0, int hi) {
  const __bf16* p = rowBase + k0 + hi * 8;
  v8bf lo  = *(const v8bf*)(p);        // 16B aligned ds_load_b128
  v8bf hi8 = *(const v8bf*)(p + 16);   // second K half
  return __builtin_shufflevector(lo, hi8, 0, 1, 2, 3, 4, 5, 6, 7,
                                          8, 9, 10, 11, 12, 13, 14, 15);
}

// Generic C(Mx384) = gather(A)(Mx128) @ W(384x128)^T (+bias), fp32 in/out,
// bf16 WMMA accumulate-fp32. Block = 64 rows of C, 256 threads = 8 waves.
// Wave w: row-tile (w&3), col-tiles [(w>>2)*12 .. +11].
__global__ __launch_bounds__(256)
void gemm_n384_k128(const float* __restrict__ A, const int* __restrict__ rowIdx,
                    int M, int aStride,
                    const float* __restrict__ W, int wStride,
                    const float* __restrict__ bias,
                    float* __restrict__ C) {
  extern __shared__ char smem[];
  __bf16* sW = (__bf16*)smem;                               // [384][LDS_STRIDE]
  __bf16* sA = (__bf16*)(smem + 384 * LDS_STRIDE * 2);      // [64][LDS_STRIDE]

  const int tid  = threadIdx.x;
  const int row0 = blockIdx.x * 64;

  // Stage the full 384x128 weight panel (bf16) into LDS.
  for (int idx = tid; idx < 384 * 128; idx += 256) {
    int g = idx >> 7, k = idx & 127;
    sW[g * LDS_STRIDE + k] = (__bf16)W[(size_t)g * wStride + k];
  }
  // Stage the (optionally gathered) 64x128 A tile (bf16) into LDS.
  for (int idx = tid; idx < 64 * 128; idx += 256) {
    int r = idx >> 7, k = idx & 127;
    int gr  = row0 + r;
    int src = (gr < M) ? (rowIdx ? rowIdx[gr] : gr) : 0;
    sA[r * LDS_STRIDE + k] = (__bf16)A[(size_t)src * aStride + k];
  }
  __syncthreads();

  const int wave = tid >> 5, lane = tid & 31;
  const int lrow = lane & 15, hi = lane >> 4;
  const int rt    = wave & 3;          // row tile (16 rows)
  const int cbase = (wave >> 2) * 12;  // first of 12 col tiles

  v8f acc[12];
#pragma unroll
  for (int j = 0; j < 12; ++j) {
    v8f z = {0.f, 0.f, 0.f, 0.f, 0.f, 0.f, 0.f, 0.f};
    acc[j] = z;
  }

  const __bf16* aRow = sA + (rt * 16 + lrow) * LDS_STRIDE;
#pragma unroll
  for (int kk = 0; kk < 4; ++kk) {
    const int k0 = kk * 32;
    v16bf a = lds_frag(aRow, k0, hi);
#pragma unroll
    for (int j = 0; j < 12; ++j) {
      const __bf16* bRow = sW + ((cbase + j) * 16 + lrow) * LDS_STRIDE;
      v16bf b = lds_frag(bRow, k0, hi);
      acc[j] = __builtin_amdgcn_wmma_f32_16x16x32_bf16(
          false, a, false, b, (short)0, acc[j], false, false);
    }
  }

  // D layout: acc[r] holds (M = rt*16 + hi*8 + r, N = cbase*16 + j*16 + lrow)
#pragma unroll
  for (int j = 0; j < 12; ++j) {
    const int n  = (cbase + j) * 16 + lrow;
    const float bn = bias ? bias[n] : 0.f;
#pragma unroll
    for (int r = 0; r < 8; ++r) {
      const int m = row0 + rt * 16 + hi * 8 + r;
      if (m < M) C[(size_t)m * 384 + n] = acc[j][r] + bn;
    }
  }
}

// GRU gates + masked mean. One block per (b,t); thread j = hidden channel.
// Gate rows are [r | z | n] stacked (torch GRUCell order), 384 wide.
__global__ __launch_bounds__(128)
void gru_reduce(const float* __restrict__ GE,   // (4096, 384) ent part of gi + b_ih
                const float* __restrict__ GR,   // (400, 384)  rel part of gi
                const float* __restrict__ GH,   // (128*258, 384) gh (incl b_hh)
                const float* __restrict__ nodeEmb,   // (128*258, 128)
                const int*   __restrict__ neighbors, // (4096, 64, 2)
                const int*   __restrict__ nnum,      // (4096)
                float*       __restrict__ upd) {     // (4096, 128)
  const int bt = blockIdx.x;
  const int b  = bt >> 5;
  const int j  = threadIdx.x;
  const int num = nnum[bt];

  const float ger = GE[(size_t)bt * 384 + j];
  const float gez = GE[(size_t)bt * 384 + 128 + j];
  const float gen = GE[(size_t)bt * 384 + 256 + j];

  float sum = 0.f;
  for (int m = 0; m < num; ++m) {
    const int node = neighbors[((size_t)bt * 64 + m) * 2 + 0];
    const int rel  = neighbors[((size_t)bt * 64 + m) * 2 + 1];
    const float* gr = GR + (size_t)rel * 384;
    const float* gh = GH + (size_t)(b * 258 + node) * 384;
    const float hv  = nodeEmb[(size_t)(b * 258 + node) * 128 + j];

    const float r = 1.f / (1.f + __expf(-(ger + gr[j]       + gh[j])));
    const float z = 1.f / (1.f + __expf(-(gez + gr[128 + j] + gh[128 + j])));
    const float n = tanhf(gen + gr[256 + j] + r * gh[256 + j]);
    sum += (1.f - z) * n + z * hv;
  }
  const float denom = (float)num + (num == 0 ? 1.f : 0.f);
  upd[(size_t)bt * 128 + j] = sum / denom;
}

__global__ void copy_f4(const float4* __restrict__ src, float4* __restrict__ dst,
                        long n4) {
  long i = (long)blockIdx.x * blockDim.x + threadIdx.x;
  const long stride = (long)gridDim.x * blockDim.x;
  for (; i < n4; i += stride) dst[i] = src[i];
}

// Deterministic scatter: one block per batch, t-loop in order -> last t wins
// on duplicate aim_nodes (matches sequential .at[].set semantics).
__global__ __launch_bounds__(128)
void scatter_upd(const float* __restrict__ upd, const int* __restrict__ aim_nodes,
                 float* __restrict__ out) {
  const int b = blockIdx.x;
  const int j = threadIdx.x;
  for (int t = 0; t < 32; ++t) {
    const int row = aim_nodes[b * 32 + t];
    out[(size_t)(b * 258 + row) * 128 + j] = upd[(size_t)(b * 32 + t) * 128 + j];
  }
}

extern "C" void kernel_launch(void* const* d_in, const int* in_sizes, int n_in,
                              void* d_out, int out_size, void* d_ws, size_t ws_size,
                              hipStream_t stream) {
  const float* node_emb = (const float*)d_in[0];   // (128, 258, 128)
  const float* ent_tab  = (const float*)d_in[1];   // (40000, 128)
  const float* rel_tab  = (const float*)d_in[2];   // (400, 128)
  const float* W_ih     = (const float*)d_in[3];   // (384, 256)
  const float* W_hh     = (const float*)d_in[4];   // (384, 128)
  const float* b_ih     = (const float*)d_in[5];   // (384)
  const float* b_hh     = (const float*)d_in[6];   // (384)
  const int*   aim_nodes = (const int*)d_in[7];    // (128, 32)
  const int*   aim_ents  = (const int*)d_in[8];    // (128, 32)
  const int*   neighbors = (const int*)d_in[9];    // (128, 32, 64, 2)
  const int*   nnum      = (const int*)d_in[10];   // (128, 32)
  float* out = (float*)d_out;

  // Workspace partition (floats): GH 33024x384 | GE 4096x384 | GR 448x384 | upd 4096x128
  float* GH  = (float*)d_ws;
  float* GE  = GH + (size_t)33024 * 384;
  float* GR  = GE + (size_t)4096 * 384;
  float* upd = GR + (size_t)448 * 384;

  const size_t shmem = (size_t)(384 + 64) * LDS_STRIDE * 2;  // ~119 KB of 320 KB LDS

  // G_h = node_embeddings @ W_hh^T + b_hh        (M = 33024)
  gemm_n384_k128<<<516, 256, shmem, stream>>>(node_emb, nullptr, 33024, 128,
                                              W_hh, 128, b_hh, GH);
  // G_ent = entity_table[aim_entities] @ W_ih[:, :128]^T + b_ih   (M = 4096)
  gemm_n384_k128<<<64, 256, shmem, stream>>>(ent_tab, aim_ents, 4096, 128,
                                             W_ih, 256, b_ih, GE);
  // G_rel = relation_table @ W_ih[:, 128:]^T     (M = 400)
  gemm_n384_k128<<<7, 256, shmem, stream>>>(rel_tab, nullptr, 400, 128,
                                            W_ih + 128, 256, nullptr, GR);

  // GRU elementwise + masked mean over neighbors.
  gru_reduce<<<4096, 128, 0, stream>>>(GE, GR, GH, node_emb, neighbors, nnum, upd);

  // out = node_embeddings, then scatter updated rows (deterministic order).
  copy_f4<<<2048, 256, 0, stream>>>((const float4*)node_emb, (float4*)out,
                                    (long)33024 * 128 / 4);
  scatter_upd<<<128, 128, 0, stream>>>(upd, aim_nodes, out);
}
